// MultiHeadAttention_29248727286121
// MI455X (gfx1250) — compile-verified, matrix operand order checked by reading, not run on hardware
//
#include <hip/hip_runtime.h>

// Problem constants (match reference)
#define BB   2
#define TT   2048
#define DM   2048
#define NH   16
#define NKV  4
#define DKH  128
#define GRP  4   // NH / NKV

typedef __attribute__((ext_vector_type(16))) __bf16 v16bf;
typedef __attribute__((ext_vector_type(8)))  __bf16 bf16x8;
typedef __attribute__((ext_vector_type(4)))  __bf16 bf16x4;
typedef __attribute__((ext_vector_type(8)))  float  v8f;

// ---------------------------------------------------------------------------
// fp32 -> bf16 round-to-nearest-even (pure ALU, no reliance on fptrunc codegen)
__device__ __forceinline__ __bf16 to_bf16(float f) {
    unsigned u = __builtin_bit_cast(unsigned, f);
    u += 0x7FFFu + ((u >> 16) & 1u);
    unsigned short s = (unsigned short)(u >> 16);
    return __builtin_bit_cast(__bf16, s);
}

// Assemble a 16x32 bf16 A-fragment (or 32x16 B-fragment) from two 8-element
// chunks. Per CDNA5 ISA layout, each lane holds K-chunks {hsel*8..+7} and
// {16+hsel*8..+7} (hsel = lane>>4), mapping to VGPRs 0-3 and 4-7.
__device__ __forceinline__ v16bf mkfrag(bf16x8 lo, bf16x8 hi) {
    v16bf f;
#pragma unroll
    for (int i = 0; i < 8; ++i) { f[i] = lo[i]; f[8 + i] = hi[i]; }
    return f;
}

__device__ __forceinline__ v16bf ldfrag_bf16(const __bf16* p, int c0, int c1) {
    return mkfrag(*(const bf16x8*)(p + c0), *(const bf16x8*)(p + c1));
}

__device__ __forceinline__ v8f wmma_bf16(v16bf a, v16bf b, v8f c) {
    return __builtin_amdgcn_wmma_f32_16x16x32_bf16(
        /*neg_a=*/false, a, /*neg_b=*/false, b,
        /*c_mod=*/(short)0, c, /*reuse_a=*/false, /*reuse_b=*/false);
}

// ---------------------------------------------------------------------------
// Bulk fp32 -> bf16 conversion, 8 elements per thread (float4 x2 in, b128 out).
__global__ __launch_bounds__(256) void cvt_f32_to_bf16(
    const float* __restrict__ in, __bf16* __restrict__ out, int n8) {
    int i = blockIdx.x * 256 + threadIdx.x;
    if (i >= n8) return;
    size_t base = (size_t)i * 8;
    float v[8];
    *(float4*)(v)     = *(const float4*)(in + base);
    *(float4*)(v + 4) = *(const float4*)(in + base + 4);
    bf16x8 o;
#pragma unroll
    for (int j = 0; j < 8; ++j) o[j] = to_bf16(v[j]);
    *(bf16x8*)(out + base) = o;
}

// ---------------------------------------------------------------------------
// C[M,N] = A[M,K] * Bw[N,K]^T   (nn.Linear: y = x @ W^T), both operands bf16.
// One wave computes a 32x64 output tile (2 A-frags x 4 B-frags = 8 WMMA per
// K-step vs 12 b128 loads), with register double-buffering so next-iteration
// loads overlap the current WMMA block.  Block = 128 threads = 4 waves (all
// share col0 -> B fragments dedup in WGP$).  grid = (N/64, M/128).
__global__ __launch_bounds__(128) void wmma_gemm_bf16(
    const __bf16* __restrict__ A, const __bf16* __restrict__ Bw,
    float* __restrict__ C, int M, int N, int K) {
    const int lane = threadIdx.x & 31;
    const int wid  = threadIdx.x >> 5;
    const int r    = lane & 15;
    const int hsel = lane >> 4;

    const int row0 = (blockIdx.y * 4 + wid) * 32;
    const int col0 = blockIdx.x * 64;

    const __bf16* arow0 = A + (size_t)(row0 + r) * K;
    const __bf16* arow1 = A + (size_t)(row0 + 16 + r) * K;
    const __bf16* brow0 = Bw + (size_t)(col0 + r) * K;
    const __bf16* brow1 = Bw + (size_t)(col0 + 16 + r) * K;
    const __bf16* brow2 = Bw + (size_t)(col0 + 32 + r) * K;
    const __bf16* brow3 = Bw + (size_t)(col0 + 48 + r) * K;

    v8f acc[2][4];
#pragma unroll
    for (int hh = 0; hh < 2; ++hh)
#pragma unroll
        for (int ct = 0; ct < 4; ++ct) acc[hh][ct] = {};

    // prologue: fragments for k0 = 0
    int c0 = hsel * 8, c1 = c0 + 16;
    v16bf a0 = ldfrag_bf16(arow0, c0, c1);
    v16bf a1 = ldfrag_bf16(arow1, c0, c1);
    v16bf b0 = ldfrag_bf16(brow0, c0, c1);
    v16bf b1 = ldfrag_bf16(brow1, c0, c1);
    v16bf b2 = ldfrag_bf16(brow2, c0, c1);
    v16bf b3 = ldfrag_bf16(brow3, c0, c1);

    for (int k0 = 32; k0 < K; k0 += 32) {
        const int n0 = k0 + hsel * 8, n1 = n0 + 16;
        // issue next-iteration loads first so they overlap the WMMA block
        v16bf na0 = ldfrag_bf16(arow0, n0, n1);
        v16bf na1 = ldfrag_bf16(arow1, n0, n1);
        v16bf nb0 = ldfrag_bf16(brow0, n0, n1);
        v16bf nb1 = ldfrag_bf16(brow1, n0, n1);
        v16bf nb2 = ldfrag_bf16(brow2, n0, n1);
        v16bf nb3 = ldfrag_bf16(brow3, n0, n1);

        acc[0][0] = wmma_bf16(a0, b0, acc[0][0]);
        acc[1][0] = wmma_bf16(a1, b0, acc[1][0]);
        acc[0][1] = wmma_bf16(a0, b1, acc[0][1]);
        acc[1][1] = wmma_bf16(a1, b1, acc[1][1]);
        acc[0][2] = wmma_bf16(a0, b2, acc[0][2]);
        acc[1][2] = wmma_bf16(a1, b2, acc[1][2]);
        acc[0][3] = wmma_bf16(a0, b3, acc[0][3]);
        acc[1][3] = wmma_bf16(a1, b3, acc[1][3]);

        a0 = na0; a1 = na1; b0 = nb0; b1 = nb1; b2 = nb2; b3 = nb3;
    }
    // epilogue K-step
    acc[0][0] = wmma_bf16(a0, b0, acc[0][0]);
    acc[1][0] = wmma_bf16(a1, b0, acc[1][0]);
    acc[0][1] = wmma_bf16(a0, b1, acc[0][1]);
    acc[1][1] = wmma_bf16(a1, b1, acc[1][1]);
    acc[0][2] = wmma_bf16(a0, b2, acc[0][2]);
    acc[1][2] = wmma_bf16(a1, b2, acc[1][2]);
    acc[0][3] = wmma_bf16(a0, b3, acc[0][3]);
    acc[1][3] = wmma_bf16(a1, b3, acc[1][3]);

#pragma unroll
    for (int hh = 0; hh < 2; ++hh)
#pragma unroll
        for (int ct = 0; ct < 4; ++ct)
#pragma unroll
            for (int i = 0; i < 8; ++i) {
                int row = row0 + hh * 16 + i + 8 * hsel;
                C[(size_t)row * N + col0 + ct * 16 + r] = acc[hh][ct][i];
            }
}

// ---------------------------------------------------------------------------
// Per-(token, head) RMSNorm over DK + interleaved-pair RoPE, fp32 in,
// bf16 out in attention layouts.  One wave per row of 128.
//   mode 0: Q  -> out[B, Hloc, T, DK]
//   mode 1: K  -> out[B, Hloc, T, DK]
//   mode 2: V  -> out[B, Hloc, DK, T]   (transposed, no norm/rope)
__global__ __launch_bounds__(256) void norm_rope_trans(
    const float* __restrict__ X, const float* __restrict__ W,
    __bf16* __restrict__ out, int Hloc, int mode) {
    const int lane = threadIdx.x & 31;
    const int g    = (int)((blockIdx.x * blockDim.x + threadIdx.x) >> 5);
    const int h    = g % Hloc;
    const int tok  = g / Hloc;
    const int b    = tok / TT;
    const int t    = tok % TT;
    const int d0   = lane * 4;

    const float* src = X + (size_t)tok * (Hloc * DKH) + h * DKH + d0;
    float x0 = src[0], x1 = src[1], x2 = src[2], x3 = src[3];

    if (mode < 2) {
        float ss = x0 * x0 + x1 * x1 + x2 * x2 + x3 * x3;
        ss += __shfl_xor(ss, 16);
        ss += __shfl_xor(ss, 8);
        ss += __shfl_xor(ss, 4);
        ss += __shfl_xor(ss, 2);
        ss += __shfl_xor(ss, 1);
        float sc = rsqrtf(ss * (1.0f / DKH) + 1.1920929e-7f);
        x0 *= sc * W[d0];     x1 *= sc * W[d0 + 1];
        x2 *= sc * W[d0 + 2]; x3 *= sc * W[d0 + 3];
        // RoPE, interleaved pairs: inv_freq = 10000^(-d_even/DK)
        const float L = 9.210340371976184f / (float)DKH;  // ln(10000)/128
        float a0 = (float)t * __expf(-(float)d0 * L);
        float a1 = (float)t * __expf(-(float)(d0 + 2) * L);
        float s0, c0, s1, c1;
        __sincosf(a0, &s0, &c0);
        __sincosf(a1, &s1, &c1);
        float n0 = x0 * c0 - x1 * s0, n1 = x1 * c0 + x0 * s0;
        float n2 = x2 * c1 - x3 * s1, n3 = x3 * c1 + x2 * s1;
        x0 = n0; x1 = n1; x2 = n2; x3 = n3;
    }

    if (mode == 2) {
        size_t base = ((size_t)(b * Hloc + h) * DKH);
        out[(base + d0 + 0) * TT + t] = to_bf16(x0);
        out[(base + d0 + 1) * TT + t] = to_bf16(x1);
        out[(base + d0 + 2) * TT + t] = to_bf16(x2);
        out[(base + d0 + 3) * TT + t] = to_bf16(x3);
    } else {
        __bf16* dst = out + (((size_t)(b * Hloc + h) * TT) + t) * DKH + d0;
        bf16x4 v;
        v[0] = to_bf16(x0); v[1] = to_bf16(x1);
        v[2] = to_bf16(x2); v[3] = to_bf16(x3);
        *(bf16x4*)dst = v;
    }
}

// ---------------------------------------------------------------------------
// Causal flash attention, GQA.  One wave owns a 16-row Q tile for one (b,h).
// Per 32-key block: 8 WMMA for Q*K^T, online softmax (fp32), P transposed
// C-layout -> A-layout through wave-private LDS, 8 WMMA for P*V.
// Block = 128 threads = 4 independent waves.
__global__ __launch_bounds__(128) void attn_fwd(
    const __bf16* __restrict__ Q, const __bf16* __restrict__ Kc,
    const __bf16* __restrict__ Vt, __bf16* __restrict__ Oc) {
    __shared__ __bf16 pls[4][16 * 32];

    const int lane = threadIdx.x & 31;
    const int wid  = threadIdx.x >> 5;
    const int r    = lane & 15;
    const int hsel = lane >> 4;

    const int gid = blockIdx.x * 4 + wid;
    const int qt  = gid & 127;         // T/16 = 128 q tiles
    const int bh  = gid >> 7;
    const int h   = bh & 15;
    const int b   = bh >> 4;
    const int hkv = h >> 2;            // repeat_interleave grouping
    const int q0  = qt * 16;

    const __bf16* qbase = Q  + ((size_t)(b * NH  + h)   * TT) * DKH;
    const __bf16* kbase = Kc + ((size_t)(b * NKV + hkv) * TT) * DKH;
    const __bf16* vbase = Vt + ((size_t)(b * NKV + hkv) * DKH) * TT;

    // Persistent Q fragments: 16 rows x 128 dk as 4 K-chunks of 32
    v16bf qf[4];
    {
        const __bf16* qrow = qbase + (size_t)(q0 + r) * DKH;
#pragma unroll
        for (int kk = 0; kk < 4; ++kk)
            qf[kk] = ldfrag_bf16(qrow, kk * 32 + hsel * 8, kk * 32 + 16 + hsel * 8);
    }

    v8f acc[8];
#pragma unroll
    for (int n = 0; n < 8; ++n) acc[n] = {};
    float m[8], l[8];
#pragma unroll
    for (int i = 0; i < 8; ++i) { m[i] = -1e30f; l[i] = 0.0f; }

    const float scale = 0.08838834764831845f;  // 1/sqrt(128)

    for (int j0 = 0; j0 < q0 + 16; j0 += 32) {
        // ---- S = Q K^T for two 16-key sub-tiles
        v8f S0 = {}, S1 = {};
        {
            const __bf16* krow0 = kbase + (size_t)(j0 + r) * DKH;
            const __bf16* krow1 = kbase + (size_t)(j0 + 16 + r) * DKH;
#pragma unroll
            for (int kk = 0; kk < 4; ++kk) {
                v16bf kf0 = ldfrag_bf16(krow0, kk * 32 + hsel * 8, kk * 32 + 16 + hsel * 8);
                S0 = wmma_bf16(qf[kk], kf0, S0);
                v16bf kf1 = ldfrag_bf16(krow1, kk * 32 + hsel * 8, kk * 32 + 16 + hsel * 8);
                S1 = wmma_bf16(qf[kk], kf1, S1);
            }
        }

        // ---- scale + causal mask (C layout: row = i + 8*hsel, col = r)
        float corr[8];
#pragma unroll
        for (int i = 0; i < 8; ++i) {
            int row = q0 + i + 8 * hsel;
            float s0 = S0[i] * scale; if (j0 + r      > row) s0 = -1e30f;
            float s1 = S1[i] * scale; if (j0 + 16 + r > row) s1 = -1e30f;

            float mx = fmaxf(s0, s1);
            mx = fmaxf(mx, __shfl_xor(mx, 1));
            mx = fmaxf(mx, __shfl_xor(mx, 2));
            mx = fmaxf(mx, __shfl_xor(mx, 4));
            mx = fmaxf(mx, __shfl_xor(mx, 8));
            float mnew = fmaxf(m[i], mx);
            float cr   = __expf(m[i] - mnew);
            float e0   = __expf(s0 - mnew);
            float e1   = __expf(s1 - mnew);
            float rs = e0 + e1;
            rs += __shfl_xor(rs, 1);
            rs += __shfl_xor(rs, 2);
            rs += __shfl_xor(rs, 4);
            rs += __shfl_xor(rs, 8);
            l[i] = l[i] * cr + rs;
            m[i] = mnew;
            corr[i] = cr;
            // stash P (bf16) into wave-private LDS tile (16 x 32)
            pls[wid][(i + 8 * hsel) * 32 + r]      = to_bf16(e0);
            pls[wid][(i + 8 * hsel) * 32 + 16 + r] = to_bf16(e1);
        }
        __builtin_amdgcn_wave_barrier();  // LDS ops are in-order per wave

        // ---- rescale running accumulators
#pragma unroll
        for (int n = 0; n < 8; ++n)
#pragma unroll
            for (int i = 0; i < 8; ++i) acc[n][i] *= corr[i];

        // ---- reload P as a 16x32 A-fragment from LDS
        const __bf16* prow = &pls[wid][r * 32];
        v16bf pf = ldfrag_bf16(prow, hsel * 8, 16 + hsel * 8);

        // ---- O += P * V   (V transposed: column dk contiguous over keys)
#pragma unroll
        for (int n = 0; n < 8; ++n) {
            const __bf16* vrow = vbase + (size_t)(n * 16 + r) * TT + j0;
            v16bf vf = ldfrag_bf16(vrow, hsel * 8, 16 + hsel * 8);
            acc[n] = wmma_bf16(pf, vf, acc[n]);
        }
    }

    // ---- epilogue: divide by l, write bf16 token-major [B*T, H*DK]
    float inv[8];
#pragma unroll
    for (int i = 0; i < 8; ++i) inv[i] = 1.0f / l[i];
#pragma unroll
    for (int n = 0; n < 8; ++n)
#pragma unroll
        for (int i = 0; i < 8; ++i) {
            int row = q0 + i + 8 * hsel;
            Oc[(size_t)(b * TT + row) * DM + h * DKH + n * 16 + r] =
                to_bf16(acc[n][i] * inv[i]);
        }
}

// ---------------------------------------------------------------------------
extern "C" void kernel_launch(void* const* d_in, const int* in_sizes, int n_in,
                              void* d_out, int out_size, void* d_ws, size_t ws_size,
                              hipStream_t stream) {
    const float* x   = (const float*)d_in[0];
    const float* wq  = (const float*)d_in[1];
    const float* wk  = (const float*)d_in[2];
    const float* wv  = (const float*)d_in[3];
    const float* wo  = (const float*)d_in[4];
    const float* qnw = (const float*)d_in[5];
    const float* knw = (const float*)d_in[6];
    float* out = (float*)d_out;

    const size_t Mtok = (size_t)BB * TT;        // 4096 tokens
    const size_t nX   = Mtok * DM;              // 8M
    const size_t nWq  = (size_t)NH * DKH * DM;  // 4M
    const size_t nWkv = (size_t)NKV * DKH * DM; // 1M
    const size_t nWo  = (size_t)DM * DM;        // 4M

    char* ws = (char*)d_ws;
    size_t off = 0;
    __bf16* xb  = (__bf16*)(ws + off); off += nX   * sizeof(__bf16);
    __bf16* wqb = (__bf16*)(ws + off); off += nWq  * sizeof(__bf16);
    __bf16* wkb = (__bf16*)(ws + off); off += nWkv * sizeof(__bf16);
    __bf16* wvb = (__bf16*)(ws + off); off += nWkv * sizeof(__bf16);
    __bf16* wob = (__bf16*)(ws + off); off += nWo  * sizeof(__bf16);
    float* qraw = (float*)(ws + off);  off += Mtok * (NH  * DKH) * sizeof(float);
    float* kraw = (float*)(ws + off);  off += Mtok * (NKV * DKH) * sizeof(float);
    float* vraw = (float*)(ws + off);  off += Mtok * (NKV * DKH) * sizeof(float);
    __bf16* qn  = (__bf16*)(ws + off); off += Mtok * (NH  * DKH) * sizeof(__bf16);
    __bf16* kn  = (__bf16*)(ws + off); off += Mtok * (NKV * DKH) * sizeof(__bf16);
    __bf16* vT  = (__bf16*)(ws + off); off += Mtok * (NKV * DKH) * sizeof(__bf16);
    __bf16* aout = (__bf16*)qraw;  // qraw is dead after stage 2; reuse for stage 3 out

    // 0) one-time down-conversion of activations + weights to bf16
    cvt_f32_to_bf16<<<(int)(nX   / 8 / 256), 256, 0, stream>>>(x,  xb,  (int)(nX / 8));
    cvt_f32_to_bf16<<<(int)(nWq  / 8 / 256), 256, 0, stream>>>(wq, wqb, (int)(nWq / 8));
    cvt_f32_to_bf16<<<(int)(nWkv / 8 / 256), 256, 0, stream>>>(wk, wkb, (int)(nWkv / 8));
    cvt_f32_to_bf16<<<(int)(nWkv / 8 / 256), 256, 0, stream>>>(wv, wvb, (int)(nWkv / 8));
    cvt_f32_to_bf16<<<(int)(nWo  / 8 / 256), 256, 0, stream>>>(wo, wob, (int)(nWo / 8));

    // 1) Q/K/V projections (bf16 WMMA, fp32 accumulate); waves own 32x64 tiles
    wmma_gemm_bf16<<<dim3(DM / 64, (int)(Mtok / 128)), 128, 0, stream>>>(
        xb, wqb, qraw, (int)Mtok, NH * DKH, DM);
    wmma_gemm_bf16<<<dim3((NKV * DKH) / 64, (int)(Mtok / 128)), 128, 0, stream>>>(
        xb, wkb, kraw, (int)Mtok, NKV * DKH, DM);
    wmma_gemm_bf16<<<dim3((NKV * DKH) / 64, (int)(Mtok / 128)), 128, 0, stream>>>(
        xb, wvb, vraw, (int)Mtok, NKV * DKH, DM);

    // 2) RMSNorm + RoPE + layout transform (one wave per row of 128)
    norm_rope_trans<<<(int)(Mtok * NH / 8), 256, 0, stream>>>(qraw, qnw, qn, NH, 0);
    norm_rope_trans<<<(int)(Mtok * NKV / 8), 256, 0, stream>>>(kraw, knw, kn, NKV, 1);
    norm_rope_trans<<<(int)(Mtok * NKV / 8), 256, 0, stream>>>(vraw, knw, vT, NKV, 2);

    // 3) Causal flash attention (B*H*(T/16) = 4096 wave-tasks, 4 waves/block)
    attn_fwd<<<(BB * NH * (TT / 16)) / 4, 128, 0, stream>>>(qn, kn, vT, aout);

    // 4) Output projection (bf16 WMMA -> fp32 out)
    wmma_gemm_bf16<<<dim3(DM / 64, (int)(Mtok / 128)), 128, 0, stream>>>(
        aout, wob, out, (int)Mtok, DM, DM);
}
